// SDFNerfNet_57483842290019
// MI455X (gfx1250) — matrix operand order repaired and checked by reading, not compiled, not run on hardware
//
#include <hip/hip_runtime.h>
#include <hip/hip_bf16.h>

typedef __attribute__((ext_vector_type(16))) _Float16 v16h;
typedef __attribute__((ext_vector_type(8)))  float    v8f;
typedef int v4i __attribute__((vector_size(16)));

#define VOXEL 0.04f
#define GRIDN 192

#if defined(__HIP_DEVICE_COMPILE__) && __has_builtin(__builtin_amdgcn_global_load_async_to_lds_b128)
#define HAVE_ASYNC 1
#else
#define HAVE_ASYNC 0
#endif

// Global / LDS address-space pointer casts (via integer to avoid AS-cast errors).
#define GAS(p) ((__attribute__((address_space(1))) v4i*)(uintptr_t)(p))
#define LAS(p) ((__attribute__((address_space(3))) v4i*)(uintptr_t)(p))

// 16-byte global -> LDS copy (async if available).
__device__ __forceinline__ void cp16(_Float16* dst_lds, const _Float16* src_g) {
#if HAVE_ASYNC
  __builtin_amdgcn_global_load_async_to_lds_b128(GAS(src_g), LAS(dst_lds), 0, 0);
#else
  uint4 t = *(const uint4*)src_g;
  *(uint4*)dst_lds = t;
#endif
}

__device__ __forceinline__ void wait_async0() {
#if HAVE_ASYNC
#if __has_builtin(__builtin_amdgcn_s_wait_asynccnt)
  __builtin_amdgcn_s_wait_asynccnt(0);
#else
  asm volatile("s_wait_asynccnt 0x0" ::: "memory");
#endif
#endif
}

// ---------------------------------------------------------------------------
// Weight conversion: f32 -> f16, transposed to [N][K], zero padded.
//   w0t: [256][32]   (K 22 -> 32)
//   w1t: [256][256]
//   w2t: [16][256]   (N 3 -> 16)
// ---------------------------------------------------------------------------
__global__ void sdfnerf_convert_weights(const float* __restrict__ w0,
                                        const float* __restrict__ w1,
                                        const float* __restrict__ w2,
                                        _Float16* __restrict__ w0t,
                                        _Float16* __restrict__ w1t,
                                        _Float16* __restrict__ w2t) {
  int id = blockIdx.x * 256 + threadIdx.x;
  if (id < 256 * 32) {
    int nn = id >> 5, k = id & 31;
    w0t[nn * 32 + k] = (_Float16)((k < 22) ? w0[k * 256 + nn] : 0.0f);
    return;
  }
  int id1 = id - 256 * 32;
  if (id1 < 256 * 256) {
    int nn = id1 >> 8, k = id1 & 255;
    w1t[nn * 256 + k] = (_Float16)w1[k * 256 + nn];
    return;
  }
  int id2 = id1 - 256 * 256;
  if (id2 < 16 * 256) {
    int nn = id2 >> 8, k = id2 & 255;
    w2t[nn * 256 + k] = (_Float16)((nn < 3) ? w2[k * 3 + nn] : 0.0f);
  }
}

// A-fragment (16x32 f16) from an LDS row-major tile (contiguous k order, same as the B loads):
// lane L holds row m=L&15; element j holds K = k0 + ((L>>4)<<4) + j
__device__ __forceinline__ v16h lds_a_frag(const _Float16* rowbase, int stride,
                                           int lane, int k0) {
  int m = lane & 15;
  int hi = (lane >> 4) & 1;
  const _Float16* p = rowbase + m * stride + k0 + hi * 16;
  union { v16h v; unsigned u[8]; } r;
#pragma unroll
  for (int q = 0; q < 4; ++q) {
    r.u[q]     = *(const unsigned*)(p + 2 * q);
    r.u[4 + q] = *(const unsigned*)(p + 8 + 2 * q);
  }
  return r.v;
}

// ---------------------------------------------------------------------------
// Fused per-ray kernel. One block = one ray (M=128 samples), 8 waves.
// All weights staged in LDS; w1 double-buffered in 32-K slabs (async).
// ---------------------------------------------------------------------------
#define XS_STRIDE 36    // 32 K + pad
#define H_STRIDE  264   // 256 N + pad
#define WS_STRIDE 40    // 32 K + pad (w0 / w1 slab rows)
#define W2_STRIDE 264   // 256 K + pad (w2 rows)

__global__ void __launch_bounds__(256)
sdfnerf_fused(const float* __restrict__ rays_o,
              const float* __restrict__ rays_d,
              const float* __restrict__ t_center,
              const float* __restrict__ vol_origin,
              const float* __restrict__ sdf_grid,
              const float* __restrict__ feat_grid,
              const float* __restrict__ ln_beta,
              const float* __restrict__ b0,
              const float* __restrict__ b1,
              const float* __restrict__ b2,
              const _Float16* __restrict__ w0t,
              const _Float16* __restrict__ w1t,
              const _Float16* __restrict__ w2t,
              float* __restrict__ out) {
  __shared__ _Float16 xs[128 * XS_STRIDE];        // input features f16
  __shared__ _Float16 h0[128 * H_STRIDE];         // activations (L0->L1->L2, reused)
  __shared__ _Float16 w0s[256 * WS_STRIDE];       // staged w0t
  __shared__ _Float16 w1s[2][256 * WS_STRIDE];    // double-buffered w1 slabs
  __shared__ _Float16 w2s[16 * W2_STRIDE];        // staged w2t
  __shared__ float scanBuf[128];
  __shared__ float tauBuf[128];
  __shared__ float rgbAcc[3];

  const int r   = blockIdx.x;
  const int tid = threadIdx.x;

  // ---- stage w0, w2 and the first w1 slab (overlaps the gather phase) ----
  {
    // w0: row n = tid, 64B per row
#pragma unroll
    for (int c = 0; c < 4; ++c)
      cp16(w0s + tid * WS_STRIDE + c * 8, w0t + tid * 32 + c * 8);
    // w2: row n = tid>>4, 32B chunk (tid&15)
    int n2 = tid >> 4, cc = tid & 15;
#pragma unroll
    for (int c = 0; c < 2; ++c)
      cp16(w2s + n2 * W2_STRIDE + cc * 16 + c * 8, w2t + n2 * 256 + cc * 16 + c * 8);
    // w1 slab kt=0: row n = tid, 64B per row
#pragma unroll
    for (int c = 0; c < 4; ++c)
      cp16(w1s[0] + tid * WS_STRIDE + c * 8, w1t + tid * 256 + 0 + c * 8);
  }

  if (tid < 3) rgbAcc[tid] = 0.0f;

  // ---------------- prep: features + sigma (threads 0..127)
  float a_s = 0.0f;
  if (tid < 128) {
    const int s = tid;
    float ox = rays_o[r * 3 + 0], oy = rays_o[r * 3 + 1], oz = rays_o[r * 3 + 2];
    float dx = rays_d[r * 3 + 0], dy = rays_d[r * 3 + 1], dz = rays_d[r * 3 + 2];
    float inv = rsqrtf(dx * dx + dy * dy + dz * dz);
    dx *= inv; dy *= inv; dz *= inv;
    float tc    = t_center[r];
    float nearb = tc - 0.1f * VOXEL;
    float farb  = tc + 0.1f * VOXEL;
    float tt = (float)s * (1.0f / 127.0f);
    float d  = nearb + (farb - nearb) * tt;
    float px = ox + dx * d, py = oy + dy * d, pz = oz + dz * d;
    float vx = vol_origin[0], vy = vol_origin[1], vz = vol_origin[2];
    int ix = (int)rintf((px - vx) * (1.0f / VOXEL));
    int iy = (int)rintf((py - vy) * (1.0f / VOXEL));
    int iz = (int)rintf((pz - vz) * (1.0f / VOXEL));
    ix = min(max(ix, 0), GRIDN - 1);
    iy = min(max(iy, 0), GRIDN - 1);
    iz = min(max(iz, 0), GRIDN - 1);
    long gi = ((long)ix * GRIDN + iy) * GRIDN + iz;
    float sdf = sdf_grid[gi];
    const float4* fp = (const float4*)(feat_grid + gi * 16);
    float4 f0 = fp[0], f1 = fp[1], f2 = fp[2], f3 = fp[3];

    _Float16* xr = xs + s * XS_STRIDE;
    xr[0] = (_Float16)px;  xr[1] = (_Float16)py;  xr[2] = (_Float16)pz;
    xr[3] = (_Float16)dx;  xr[4] = (_Float16)dy;  xr[5] = (_Float16)dz;
    xr[6]  = (_Float16)f0.x; xr[7]  = (_Float16)f0.y; xr[8]  = (_Float16)f0.z; xr[9]  = (_Float16)f0.w;
    xr[10] = (_Float16)f1.x; xr[11] = (_Float16)f1.y; xr[12] = (_Float16)f1.z; xr[13] = (_Float16)f1.w;
    xr[14] = (_Float16)f2.x; xr[15] = (_Float16)f2.y; xr[16] = (_Float16)f2.z; xr[17] = (_Float16)f2.w;
    xr[18] = (_Float16)f3.x; xr[19] = (_Float16)f3.y; xr[20] = (_Float16)f3.z; xr[21] = (_Float16)f3.w;
#pragma unroll
    for (int j = 22; j < 32; ++j) xr[j] = (_Float16)0.0f;

    float beta  = __expf(ln_beta[0] * 10.0f);
    float alpha = 1.0f / beta;
    float psi   = (sdf <= 0.0f) ? (1.0f - 0.5f * __expf(sdf / beta))
                                : (0.5f * __expf(-sdf / beta));
    float sigma = alpha * psi;
    float delta = (farb - nearb) * (1.0f / 127.0f);
    a_s = -fmaxf(sigma * delta, 0.0f);   // log(p_i)
    scanBuf[tid] = a_s;
  }
  __syncthreads();

  // ---------------- cumprod via log-space scan
  for (int off = 1; off < 128; off <<= 1) {
    float t = 0.0f;
    if (tid < 128 && tid >= off) t = scanBuf[tid - off];
    __syncthreads();
    if (tid < 128) scanBuf[tid] += t;
    __syncthreads();
  }
  if (tid < 128) {
    float excl  = scanBuf[tid] - a_s;
    float trans = __expf(excl);
    float p     = __expf(a_s);
    tauBuf[tid] = (tid < 127) ? ((1.0f - p + 1e-10f) * trans) : 0.0f;
  }
  wait_async0();           // w0s / w2s / w1 slab0 in LDS
  __syncthreads();         // xs, tau and staged weights visible to all waves

  // ---------------- WMMA MLP: wave owns rows [16w, 16w+16)
  const int wv   = tid >> 5;
  const int lane = tid & 31;
  const int n    = lane & 15;
  const int hi   = (lane >> 4) & 1;
  _Float16* hrow = h0 + wv * 16 * H_STRIDE;

  // Layer 0: [16x32] x [32x256], B from LDS
  v16h a0 = lds_a_frag(xs + wv * 16 * XS_STRIDE, XS_STRIDE, lane, 0);
#pragma unroll
  for (int nt = 0; nt < 16; ++nt) {
    v16h b = *(const v16h*)(w0s + (nt * 16 + n) * WS_STRIDE + hi * 16);
    v8f c = {};
    c = __builtin_amdgcn_wmma_f32_16x16x32_f16(false, a0, false, b,
                                               (short)0, c, false, false);
    float bias = b0[nt * 16 + n];
#pragma unroll
    for (int v = 0; v < 8; ++v) {
      int m = v + 8 * hi;
      hrow[m * H_STRIDE + nt * 16 + n] = (_Float16)fmaxf(c[v] + bias, 0.0f);
    }
  }

  // Layer 1: kt-outer, double-buffered w1 slabs, 16 live acc tiles.
  v8f acc[16];
#pragma unroll
  for (int nt = 0; nt < 16; ++nt) acc[nt] = (v8f){};

  for (int kt = 0; kt < 8; ++kt) {
    if (kt < 7) {          // prefetch next slab while computing this one
      _Float16* dst = w1s[(kt + 1) & 1];
      const _Float16* src = w1t + tid * 256 + (kt + 1) * 32;
#pragma unroll
      for (int c = 0; c < 4; ++c)
        cp16(dst + tid * WS_STRIDE + c * 8, src + c * 8);
    }
    v16h a = lds_a_frag(hrow, H_STRIDE, lane, kt * 32);
    const _Float16* sl = w1s[kt & 1];
#pragma unroll
    for (int nt = 0; nt < 16; ++nt) {
      v16h b = *(const v16h*)(sl + (nt * 16 + n) * WS_STRIDE + hi * 16);
      acc[nt] = __builtin_amdgcn_wmma_f32_16x16x32_f16(false, a, false, b,
                                                       (short)0, acc[nt],
                                                       false, false);
    }
    if (kt < 7) {
      wait_async0();       // next slab landed
      __syncthreads();     // all waves done reading current slab + next visible
    }
  }
  // bias + relu + write H1 back over h0 (own rows only)
#pragma unroll
  for (int nt = 0; nt < 16; ++nt) {
    float bias = b1[nt * 16 + n];
#pragma unroll
    for (int v = 0; v < 8; ++v) {
      int m = v + 8 * hi;
      hrow[m * H_STRIDE + nt * 16 + n] = (_Float16)fmaxf(acc[nt][v] + bias, 0.0f);
    }
  }

  // Layer 2: [16x256] x [256x16] (N padded 3->16), B from LDS
  v8f c2 = {};
#pragma unroll
  for (int kt = 0; kt < 8; ++kt) {
    v16h a = lds_a_frag(hrow, H_STRIDE, lane, kt * 32);
    v16h b = *(const v16h*)(w2s + n * W2_STRIDE + kt * 32 + hi * 16);
    c2 = __builtin_amdgcn_wmma_f32_16x16x32_f16(false, a, false, b,
                                                (short)0, c2, false, false);
  }
  float bias2 = (n < 3) ? b2[n] : 0.0f;
  float partial = 0.0f;
#pragma unroll
  for (int v = 0; v < 8; ++v) {
    int row = wv * 16 + v + 8 * hi;
    float x = c2[v] + bias2;
    float sg = 1.0f / (1.0f + __expf(-x));
    partial += sg * tauBuf[row];
  }
  if (n < 3) atomicAdd(&rgbAcc[n], partial);
  __syncthreads();
  if (tid < 3) out[r * 3 + tid] = rgbAcc[tid];
}

// ---------------------------------------------------------------------------
extern "C" void kernel_launch(void* const* d_in, const int* in_sizes, int n_in,
                              void* d_out, int out_size, void* d_ws, size_t ws_size,
                              hipStream_t stream) {
  const float* rays_o     = (const float*)d_in[0];
  const float* rays_d     = (const float*)d_in[1];
  const float* t_center   = (const float*)d_in[2];
  const float* vol_origin = (const float*)d_in[3];
  const float* sdf_grid   = (const float*)d_in[4];
  const float* feat_grid  = (const float*)d_in[5];
  const float* ln_beta    = (const float*)d_in[6];
  const float* w0 = (const float*)d_in[7];
  const float* b0 = (const float*)d_in[8];
  const float* w1 = (const float*)d_in[9];
  const float* b1 = (const float*)d_in[10];
  const float* w2 = (const float*)d_in[11];
  const float* b2 = (const float*)d_in[12];

  _Float16* w0t = (_Float16*)d_ws;                              // [256][32]
  _Float16* w1t = (_Float16*)((char*)d_ws + 16384);             // [256][256]
  _Float16* w2t = (_Float16*)((char*)d_ws + 16384 + 131072);    // [16][256]

  int conv_elems = 256 * 32 + 256 * 256 + 16 * 256;             // 77824
  sdfnerf_convert_weights<<<(conv_elems + 255) / 256, 256, 0, stream>>>(
      w0, w1, w2, w0t, w1t, w2t);

  sdfnerf_fused<<<4096, 256, 0, stream>>>(
      rays_o, rays_d, t_center, vol_origin, sdf_grid, feat_grid, ln_beta,
      b0, b1, b2, w0t, w1t, w2t, (float*)d_out);
}
